// LSTMModule_72103910965344
// MI455X (gfx1250) — matrix operand order repaired and checked by reading, not compiled, not run on hardware
//
#include <hip/hip_runtime.h>
#include <hip/hip_bf16.h>

// ---------------------------------------------------------------------------
// Bidirectional 2-layer LSTM for MI455X (gfx1250, wave32, WMMA).
//
// Decomposition:
//   * conversion kernels: f32 weights -> bf16 (layer0 Wih zero-padded K=4->32),
//     bias pairs combined.
//   * lstm_dir_kernel<LAYER>: grid (B/16, 2 dirs), block = 1 wave (32 lanes).
//     Each wave owns a 16-row batch tile and walks all 1000 time steps.
//     Per step: gates[16x128] = in@Wih^T + h@Whh^T + bias via
//     v_wmma_f32_16x16x32_bf16 (8 N-tiles x up-to-3 K-frags), then fused
//     sigmoid/tanh gate math, c/h update; h round-trips through LDS to
//     re-shape D-tile -> A-fragment for the next step's recurrent WMMA.
//   * head_kernel: features = sigmoid(rnn@help_W^T+b); out = (feat*x)@out_W^T.
//
// Workspace: ~262.3 MB (two bf16 [B,L,64] buffers + bf16 weights + biases).
// ---------------------------------------------------------------------------

typedef __bf16 bf16_t;
typedef __attribute__((ext_vector_type(16))) __bf16 v16bf;
typedef __attribute__((ext_vector_type(8)))  __bf16 v8bf;
typedef __attribute__((ext_vector_type(8)))  float  v8f;

#define B_   1024
#define C_   4
#define L_   1000
#define H_   32
#define NC_  2
#define G4H  128   // 4*H (gate width)

__device__ __forceinline__ float sigmf(float x) { return 1.0f / (1.0f + __expf(-x)); }

// A-fragment (16x32 bf16, M x K). rowbase = start of this lane's matrix row
// (row = lane & 15). Element i -> K = koff + 16*(i/8) + hi8 + (i%8),
// hi8 = 8*(lane>=16). Two aligned 16B loads per fragment.
__device__ __forceinline__ v16bf load_a_frag(const bf16_t* rowbase, int koff, int hi8) {
    const v8bf lo = *(const v8bf*)(rowbase + koff + hi8);
    const v8bf hi = *(const v8bf*)(rowbase + koff + hi8 + 16);
    return __builtin_shufflevector(lo, hi, 0, 1, 2, 3, 4, 5, 6, 7,
                                           8, 9, 10, 11, 12, 13, 14, 15);
}

// B-fragment (32x16 bf16, K x N) for gates = in @ W^T, i.e. B[k][n] = W[n][k].
// W row-major [128][ldk]. Element i -> K = k0 + 16*(i/8) + 8*(lane>=16) + (i%8), N = n0 + lane%16.
__device__ __forceinline__ v16bf load_b_frag(const bf16_t* W, int ldk, int n0, int k0, int lane) {
    const int n  = n0 + (lane & 15);
    const int kh = k0 + ((lane >> 4) << 3);
    const bf16_t* r = W + n * ldk + kh;
    const v8bf lo = *(const v8bf*)(r);
    const v8bf hi = *(const v8bf*)(r + 16);
    return __builtin_shufflevector(lo, hi, 0, 1, 2, 3, 4, 5, 6, 7,
                                           8, 9, 10, 11, 12, 13, 14, 15);
}

// ---------------------------------------------------------------------------
// Weight conversion kernels
// ---------------------------------------------------------------------------
__global__ void cvt_bf16_kernel(bf16_t* __restrict__ dst, const float* __restrict__ src, int n) {
    int i = blockIdx.x * 256 + threadIdx.x;
    if (i < n) dst[i] = (bf16_t)src[i];
}

// Pad [rows][sc] f32 -> [rows][dc] bf16 with zeros in columns sc..dc-1.
__global__ void cvt_pad_kernel(bf16_t* __restrict__ dst, const float* __restrict__ src,
                               int rows, int sc, int dc) {
    int i = blockIdx.x * 256 + threadIdx.x;
    if (i < rows * dc) {
        int r = i / dc, c = i % dc;
        dst[i] = (c < sc) ? (bf16_t)src[r * sc + c] : (bf16_t)0.0f;
    }
}

__global__ void bias_kernel(float* __restrict__ dst, const float* __restrict__ a,
                            const float* __restrict__ b, int n) {
    int i = blockIdx.x * 256 + threadIdx.x;
    if (i < n) dst[i] = a[i] + b[i];
}

// ---------------------------------------------------------------------------
// Recurrent LSTM kernel: one wave per (16-batch-row tile, direction).
// LAYER 0: input from x [B,C,L] (K padded 4->32).  LAYER 1: input l0 [B,L,64].
// Output written bf16 into out[B,L,64] at column offset dir*32 (fwd/bwd concat).
// ---------------------------------------------------------------------------
template <int LAYER>
__global__ void lstm_dir_kernel(const float*  __restrict__ x,
                                const bf16_t* __restrict__ l0,
                                const bf16_t* __restrict__ WihG,   // [2][128*KW]
                                const bf16_t* __restrict__ WhhG,   // [2][128*32]
                                const float*  __restrict__ biasG,  // [2][128]
                                bf16_t*       __restrict__ out) {  // [B,L,64]
    constexpr int KW = (LAYER == 0) ? 32 : 64;

    const int lane = threadIdx.x;           // 0..31, single wave per block
    const int b0   = blockIdx.x * 16;
    const int dir  = blockIdx.y;            // 0 = forward, 1 = backward

    __shared__ __align__(16) bf16_t sWih[G4H * KW];
    __shared__ __align__(16) bf16_t sWhh[G4H * 32];
    __shared__ __align__(16) float  sBias[G4H];
    __shared__ __align__(16) bf16_t sH[16 * 32];

    // Stage this direction's weights into LDS (16B vector copies).
    {
        const uint4* s = (const uint4*)(WihG + (size_t)dir * G4H * KW);
        uint4* d = (uint4*)sWih;
        for (int i = lane; i < G4H * KW / 8; i += 32) d[i] = s[i];
    }
    {
        const uint4* s = (const uint4*)(WhhG + (size_t)dir * G4H * 32);
        uint4* d = (uint4*)sWhh;
        for (int i = lane; i < G4H * 32 / 8; i += 32) d[i] = s[i];
    }
    for (int i = lane; i < G4H; i += 32)   sBias[i] = biasG[dir * G4H + i];
    for (int i = lane; i < 16 * 32; i += 32) sH[i] = (bf16_t)0.0f;
    __syncthreads();

    const int row  = lane & 15;             // batch row within tile / A-frag M
    const int hi8  = (lane >> 4) << 3;      // A-frag half-wave K shift
    const int m0   = (lane >> 4) << 3;      // D-tile row base
    const int ncol = lane & 15;             // D-tile column
    const size_t bRow = (size_t)(b0 + row);

    v8f cst[2];                             // cell state c, 2 x (16x16) f32 tiles
    cst[0] = (v8f)0.0f;
    cst[1] = (v8f)0.0f;

    for (int s = 0; s < L_; ++s) {
        const int t = dir ? (L_ - 1 - s) : s;

        // ---- input A fragments -------------------------------------------
        v16bf a_in0, a_in1;
        if constexpr (LAYER == 0) {
            v16bf z = {};
            if (lane < 16) {                 // only K=0..3 are non-zero
                z[0] = (bf16_t)x[(bRow * 4 + 0) * L_ + t];
                z[1] = (bf16_t)x[(bRow * 4 + 1) * L_ + t];
                z[2] = (bf16_t)x[(bRow * 4 + 2) * L_ + t];
                z[3] = (bf16_t)x[(bRow * 4 + 3) * L_ + t];
            }
            a_in0 = z;
            a_in1 = z;                       // unused
        } else {
            const bf16_t* rb = l0 + (bRow * L_ + (size_t)t) * 64;
            a_in0 = load_a_frag(rb, 0, hi8);
            a_in1 = load_a_frag(rb, 32, hi8);
        }
        // recurrent h fragment from LDS (previous step's h, bf16 [16][32])
        const v16bf a_h = load_a_frag(sH + row * 32, 0, hi8);

        // ---- gates[16x128]: 8 N-tiles of f32 WMMA accumulators ------------
        v8f g[8];
#pragma unroll
        for (int nt = 0; nt < 8; ++nt) {
            const float bv = sBias[nt * 16 + ncol];
            v8f acc = {bv, bv, bv, bv, bv, bv, bv, bv};
            acc = __builtin_amdgcn_wmma_f32_16x16x32_bf16(
                false, a_in0, false, load_b_frag(sWih, KW, nt * 16, 0, lane),
                (short)0, acc, false, false);
            if constexpr (LAYER == 1) {
                acc = __builtin_amdgcn_wmma_f32_16x16x32_bf16(
                    false, a_in1, false, load_b_frag(sWih, KW, nt * 16, 32, lane),
                    (short)0, acc, false, false);
            }
            acc = __builtin_amdgcn_wmma_f32_16x16x32_bf16(
                false, a_h, false, load_b_frag(sWhh, 32, nt * 16, 0, lane),
                (short)0, acc, false, false);
            g[nt] = acc;
        }

        // ---- gate activations, c/h update; new h -> LDS -------------------
        // PyTorch gate order: tiles 0,1=i  2,3=f  4,5=g  6,7=o
#pragma unroll
        for (int j = 0; j < 2; ++j) {
            const v8f iv = g[0 + j], fv = g[2 + j], gv = g[4 + j], ov = g[6 + j];
#pragma unroll
            for (int e = 0; e < 8; ++e) {
                float cNew = sigmf(fv[e]) * cst[j][e] + sigmf(iv[e]) * tanhf(gv[e]);
                cst[j][e] = cNew;
                const float hh = sigmf(ov[e]) * tanhf(cNew);
                sH[(m0 + e) * 32 + j * 16 + ncol] = (bf16_t)hh;
            }
        }
        __syncthreads();   // single wave: orders LDS writes vs. vector reads below

        // ---- stream h tile to global out (bf16, concat column dir*32) -----
        {
            const int r2   = lane >> 1;
            const int half = (lane & 1) * 16;
            const v8bf h0 = *(const v8bf*)(sH + r2 * 32 + half);
            const v8bf h1 = *(const v8bf*)(sH + r2 * 32 + half + 8);
            bf16_t* dst = out + ((size_t)(b0 + r2) * L_ + t) * 64 + dir * 32 + half;
            *(v8bf*)dst       = h0;
            *(v8bf*)(dst + 8) = h1;
        }
    }
}

// ---------------------------------------------------------------------------
// Head: features = sigmoid(rnn @ help_W^T + help_b); focus = features * x;
// out = focus.reshape(B, L*C) @ out_W^T + out_b.   One block (64 thr) per batch.
// ---------------------------------------------------------------------------
__global__ void head_kernel(const bf16_t* __restrict__ rnn,    // [B,L,64]
                            const float*  __restrict__ x,      // [B,C,L]
                            const float*  __restrict__ helpW,  // [4,64]
                            const float*  __restrict__ helpB,  // [4]
                            const float*  __restrict__ outW,   // [2, L*C]
                            const float*  __restrict__ outB,   // [2]
                            float*        __restrict__ outp) { // [B,2]
    const int t = threadIdx.x;   // 0..63 : hidden index
    const int b = blockIdx.x;

    __shared__ float red[4][64];
    __shared__ float accs[4][2];

    const float hw0 = helpW[0 * 64 + t];
    const float hw1 = helpW[1 * 64 + t];
    const float hw2 = helpW[2 * 64 + t];
    const float hw3 = helpW[3 * 64 + t];

    float a0 = 0.0f, a1 = 0.0f;           // per-c partials, threads 0..3
    const bf16_t* rp = rnn + (size_t)b * L_ * 64;

    for (int l = 0; l < L_; ++l) {
        const float v = (float)rp[l * 64 + t];
        red[0][t] = v * hw0;
        red[1][t] = v * hw1;
        red[2][t] = v * hw2;
        red[3][t] = v * hw3;
        __syncthreads();
        if (t < 4) {
            float s = 0.0f;
#pragma unroll 8
            for (int i = 0; i < 64; ++i) s += red[t][i];
            const float feat  = sigmf(s + helpB[t]);
            const float focus = feat * x[((size_t)b * 4 + t) * L_ + l];
            a0 += focus * outW[0 * (L_ * C_) + l * 4 + t];
            a1 += focus * outW[1 * (L_ * C_) + l * 4 + t];
        }
        __syncthreads();
    }
    if (t < 4) { accs[t][0] = a0; accs[t][1] = a1; }
    __syncthreads();
    if (t < 2) {
        float s = outB[t];
        for (int c = 0; c < 4; ++c) s += accs[c][t];
        outp[b * 2 + t] = s;
    }
}

// ---------------------------------------------------------------------------
extern "C" void kernel_launch(void* const* d_in, const int* in_sizes, int n_in,
                              void* d_out, int out_size, void* d_ws, size_t ws_size,
                              hipStream_t stream) {
    (void)in_sizes; (void)n_in; (void)out_size; (void)ws_size;

    const float* x = (const float*)d_in[0];
    // dict order: x, then (Wih,Whh,bih,bhh) for l0f,l0b,l1f,l1b, then head
    const float* Wih_l0[2] = {(const float*)d_in[1],  (const float*)d_in[5]};
    const float* Whh_l0[2] = {(const float*)d_in[2],  (const float*)d_in[6]};
    const float* bih_l0[2] = {(const float*)d_in[3],  (const float*)d_in[7]};
    const float* bhh_l0[2] = {(const float*)d_in[4],  (const float*)d_in[8]};
    const float* Wih_l1[2] = {(const float*)d_in[9],  (const float*)d_in[13]};
    const float* Whh_l1[2] = {(const float*)d_in[10], (const float*)d_in[14]};
    const float* bih_l1[2] = {(const float*)d_in[11], (const float*)d_in[15]};
    const float* bhh_l1[2] = {(const float*)d_in[12], (const float*)d_in[16]};
    const float* helpW = (const float*)d_in[17];
    const float* helpB = (const float*)d_in[18];
    const float* outW  = (const float*)d_in[19];
    const float* outB  = (const float*)d_in[20];

    char* ws = (char*)d_ws;
    size_t off = 0;
    auto take = [&](size_t bytes) -> char* {
        char* p = ws + off;
        off = (off + bytes + 255) & ~(size_t)255;
        return p;
    };

    bf16_t* wih0 = (bf16_t*)take((size_t)2 * G4H * 32 * sizeof(bf16_t)); // padded K=32
    bf16_t* wih1 = (bf16_t*)take((size_t)2 * G4H * 64 * sizeof(bf16_t));
    bf16_t* whh0 = (bf16_t*)take((size_t)2 * G4H * 32 * sizeof(bf16_t));
    bf16_t* whh1 = (bf16_t*)take((size_t)2 * G4H * 32 * sizeof(bf16_t));
    float*  bia0 = (float*)take((size_t)2 * G4H * sizeof(float));
    float*  bia1 = (float*)take((size_t)2 * G4H * sizeof(float));
    bf16_t* l0b  = (bf16_t*)take((size_t)B_ * L_ * 64 * sizeof(bf16_t)); // 131 MB
    bf16_t* rnnb = (bf16_t*)take((size_t)B_ * L_ * 64 * sizeof(bf16_t)); // 131 MB

    // ---- weight conversions (f32 -> bf16, pad layer0 Wih, combine biases) --
    const int n0 = G4H * 32, n1 = G4H * 64;
    for (int d = 0; d < 2; ++d) {
        cvt_pad_kernel<<<(n0 + 255) / 256, 256, 0, stream>>>(wih0 + (size_t)d * n0, Wih_l0[d], G4H, 4, 32);
        cvt_bf16_kernel<<<(n1 + 255) / 256, 256, 0, stream>>>(wih1 + (size_t)d * n1, Wih_l1[d], n1);
        cvt_bf16_kernel<<<(n0 + 255) / 256, 256, 0, stream>>>(whh0 + (size_t)d * n0, Whh_l0[d], n0);
        cvt_bf16_kernel<<<(n0 + 255) / 256, 256, 0, stream>>>(whh1 + (size_t)d * n0, Whh_l1[d], n0);
        bias_kernel<<<1, 128, 0, stream>>>(bia0 + d * G4H, bih_l0[d], bhh_l0[d], G4H);
        bias_kernel<<<1, 128, 0, stream>>>(bia1 + d * G4H, bih_l1[d], bhh_l1[d], G4H);
    }

    // ---- recurrent layers (stream order gives layer0 -> layer1 dependency) -
    dim3 grid(B_ / 16, 2);
    lstm_dir_kernel<0><<<grid, 32, 0, stream>>>(x, nullptr, wih0, whh0, bia0, l0b);
    lstm_dir_kernel<1><<<grid, 32, 0, stream>>>(nullptr, l0b, wih1, whh1, bia1, rnnb);

    // ---- head --------------------------------------------------------------
    head_kernel<<<B_, 64, 0, stream>>>(rnnb, x, helpW, helpB, outW, outB, (float*)d_out);
}
